// ScaledDotProductAttention_23321672418019
// MI455X (gfx1250) — matrix operand order repaired and checked by reading, not compiled, not run on hardware
//
#include <hip/hip_runtime.h>

typedef __attribute__((ext_vector_type(16))) _Float16 v16h;
typedef __attribute__((ext_vector_type(8)))  _Float16 v8h;
typedef __attribute__((ext_vector_type(8)))  float    v8f;

// Problem constants (reference: B=4, H=16, S=2048, D=64, fp32 in/out)
constexpr int S  = 2048;
constexpr int D  = 64;
constexpr int KB = 64;   // keys processed per main-loop iteration
// Block = 256 threads = 8 waves; each wave owns 32 queries (2x 16-row subtiles)
// => 256 queries per block, one (b,h) per blockIdx.x, grid.y = S/256.

// Fragment layout assumptions (CDNA5 ISA, 16-bit, wave32):
//  A (16xK, lane&15 = M row):  v16h elem (2v+h): K = (v>=4)*16 + hi*8 + (v&3)*2 + h
//  B (Kx16, lane&15 = N col):  v16h elem (2v+h): K as for A (S^T pass: K = hi*16 + 2v + h in both A and B)
//  C (16x16 f32, lane&15 = N): v8f  elem v     : M = v + hi*8
//
// Scores are computed transposed (S^T = K_tile x Q^T) so lane&15 == query in
// the score C-fragment: softmax stats are per-lane scalars (one shfl_xor(16)
// merges lane halves) and P feeds the P*V A-fragment via per-lane repack only.
//
// Softmax runs in the base-2 domain: (1/sqrt(D))*log2(e) is folded into the
// one-time Q->f16 conversion, so exp() is a bare v_exp_f32 (no per-element
// multiply) and the result is mathematically identical to base-e softmax.

#if __has_builtin(__builtin_amdgcn_exp2f)
#define EXP2F(x) __builtin_amdgcn_exp2f(x)
#else
#define EXP2F(x) exp2f(x)
#endif

__device__ __forceinline__ v16h cat16(v8h lo, v8h hh) {
  return __builtin_shufflevector(lo, hh, 0, 1, 2, 3, 4, 5, 6, 7,
                                 8, 9, 10, 11, 12, 13, 14, 15);
}

__global__ __launch_bounds__(256)
void fa_fwd_f16wmma(const float* __restrict__ Qg, const float* __restrict__ Kg,
                    const float* __restrict__ Vg, float* __restrict__ Og) {
  // Double-buffered f16 tiles: K as [key][d], V transposed as [d][key].
  __shared__ __align__(16) _Float16 ktile[2][KB * D];  // 2 x 8 KB
  __shared__ __align__(16) _Float16 vtile[2][D * KB];  // 2 x 8 KB

  const int lane = threadIdx.x & 31;
  const int wave = threadIdx.x >> 5;
  const int n16  = lane & 15;
  const int hi   = lane >> 4;

  const int bh = blockIdx.x;                    // 0..B*H-1
  const int q0 = blockIdx.y * 256 + wave * 32;  // this wave's 32-query span

  const float* q_ptr = Qg + (size_t)bh * S * D;
  const float* k_ptr = Kg + (size_t)bh * S * D;
  const float* v_ptr = Vg + (size_t)bh * S * D;
  float*       o_ptr = Og + (size_t)bh * S * D;

  // ---- Q^T B-fragments; (1/sqrt(D)) * log2(e) folded in, loaded once ----
  const float qscale = 0.125f * 1.44269504088896340736f;
  v16h qb[2][2];  // [subtile][d-chunk]
#pragma unroll
  for (int u = 0; u < 2; ++u) {
    const float* qrow = q_ptr + (size_t)(q0 + u * 16 + n16) * D;
#pragma unroll
    for (int c = 0; c < 2; ++c)
#pragma unroll
      for (int v = 0; v < 8; ++v) {
        const int d = c * 32 + hi * 16 + v * 2;
        qb[u][c][2 * v]     = (_Float16)(qrow[d] * qscale);
        qb[u][c][2 * v + 1] = (_Float16)(qrow[d + 1] * qscale);
      }
  }

  // ---- accumulators / online-softmax state (log2 domain) ----
  v8f acc[2][4];
#pragma unroll
  for (int u = 0; u < 2; ++u)
#pragma unroll
    for (int t = 0; t < 4; ++t)
#pragma unroll
      for (int i = 0; i < 8; ++i) acc[u][t][i] = 0.0f;
  float m_i[2] = {-__builtin_inff(), -__builtin_inff()};
  float l_i[2] = {0.0f, 0.0f};

  // ---- cooperative f32->f16 staging of one 64-key K/V tile ----
  auto stage = [&](int kb, int pb) {
    {  // K tile: [key][d]; thread -> (key = tid/4, 16 d's)
      const int k  = threadIdx.x >> 2;
      const int d0 = (threadIdx.x & 3) * 16;
      const float* src = k_ptr + (size_t)(kb + k) * D + d0;
      float tmp[16];
#pragma unroll
      for (int i = 0; i < 4; ++i) {
        const float4 f = ((const float4*)src)[i];
        tmp[4 * i + 0] = f.x; tmp[4 * i + 1] = f.y;
        tmp[4 * i + 2] = f.z; tmp[4 * i + 3] = f.w;
      }
      v8h h0, h1;
#pragma unroll
      for (int i = 0; i < 8; ++i) {
        h0[i] = (_Float16)tmp[i];
        h1[i] = (_Float16)tmp[8 + i];
      }
      *(v8h*)&ktile[pb][k * D + d0]     = h0;
      *(v8h*)&ktile[pb][k * D + d0 + 8] = h1;
    }
    {  // V tile transposed: [d][key]; thread -> (d = tid%64, 16 keys)
      const int d  = threadIdx.x & 63;
      const int k0 = (threadIdx.x >> 6) * 16;
      float tmp[16];
#pragma unroll
      for (int i = 0; i < 16; ++i)
        tmp[i] = v_ptr[(size_t)(kb + k0 + i) * D + d];
      v8h h0, h1;
#pragma unroll
      for (int i = 0; i < 8; ++i) {
        h0[i] = (_Float16)tmp[i];
        h1[i] = (_Float16)tmp[8 + i];
      }
      *(v8h*)&vtile[pb][d * KB + k0]     = h0;
      *(v8h*)&vtile[pb][d * KB + k0 + 8] = h1;
    }
  };

  stage(0, 0);
  __syncthreads();

  int p = 0;
  for (int kb = 0; kb < S; kb += KB) {
    const _Float16* kt = ktile[p];
    const _Float16* vt = vtile[p];

#pragma unroll
    for (int u = 0; u < 2; ++u) {
      // ---- S^T: 4 key-tiles of 16, contraction over d in 2 chunks ----
      v8f s[4];
#pragma unroll
      for (int t = 0; t < 4; ++t)
#pragma unroll
        for (int i = 0; i < 8; ++i) s[t][i] = 0.0f;

#pragma unroll
      for (int t = 0; t < 4; ++t) {
        const _Float16* krow = kt + (t * 16 + n16) * D;  // lane&15 = key (A's M)
#pragma unroll
        for (int c = 0; c < 2; ++c) {
          const _Float16* ab = krow + c * 32 + hi * 16;
          const v16h ka = cat16(*(const v8h*)ab, *(const v8h*)(ab + 8));
          s[t] = __builtin_amdgcn_wmma_f32_16x16x32_f16(
              false, ka, false, qb[u][c], (short)0, s[t], false, false);
        }
      }

      // ---- online softmax (row = query = n16; keys over vgprs+hi+tiles) ----
      float mx = -__builtin_inff();
#pragma unroll
      for (int t = 0; t < 4; ++t)
#pragma unroll
        for (int v = 0; v < 8; ++v) mx = fmaxf(mx, s[t][v]);
      mx = fmaxf(mx, __shfl_xor(mx, 16, 32));

      const float m_new = fmaxf(m_i[u], mx);
      const float alpha = EXP2F(m_i[u] - m_new);  // 0 on first block
      float ts = 0.0f;
#pragma unroll
      for (int t = 0; t < 4; ++t)
#pragma unroll
        for (int v = 0; v < 8; ++v) {
          s[t][v] = EXP2F(s[t][v] - m_new);  // s now holds P
          ts += s[t][v];
        }
      ts += __shfl_xor(ts, 16, 32);
      l_i[u] = l_i[u] * alpha + ts;
      m_i[u] = m_new;

      // rescale output accumulators: row of acc = v + hi*8
#pragma unroll
      for (int v = 0; v < 8; ++v) {
        const float a_r = __shfl(alpha, v + hi * 8, 32);
#pragma unroll
        for (int t2 = 0; t2 < 4; ++t2) acc[u][t2][v] *= a_r;
      }

      // ---- P -> A-fragments (pure per-lane repack), 2 key-chunks of 32 ----
      v16h pa[2];
#pragma unroll
      for (int g = 0; g < 2; ++g)
#pragma unroll
        for (int v = 0; v < 8; ++v) {
          const int src = g * 2 + ((v >= 4) ? 1 : 0);
          const int w   = (v & 3) * 2;
          pa[g][2 * v]     = (_Float16)s[src][w];
          pa[g][2 * v + 1] = (_Float16)s[src][w + 1];
        }

      // ---- O += P(16x64) x V(64x64): 4 d-tiles x 2 key-chunks ----
#pragma unroll
      for (int t2 = 0; t2 < 4; ++t2) {
        const _Float16* vcol = vt + (t2 * 16 + n16) * KB;  // lane&15 = d col
#pragma unroll
        for (int g = 0; g < 2; ++g) {
          const _Float16* bb = vcol + g * 32 + hi * 8;
          const v16h vb = cat16(*(const v8h*)bb, *(const v8h*)(bb + 16));
          acc[u][t2] = __builtin_amdgcn_wmma_f32_16x16x32_f16(
              false, pa[g], false, vb, (short)0, acc[u][t2], false, false);
        }
      }
    }

    // overlap next tile's staging with this iteration's tail, 1 barrier/iter
    if (kb + KB < S) stage(kb + KB, p ^ 1);
    __syncthreads();
    p ^= 1;
  }

  // ---- finalize: reciprocal of row sum, broadcast, store fp32 output ----
#pragma unroll
  for (int u = 0; u < 2; ++u) {
    const float invl = 1.0f / l_i[u];  // per-lane (row = n16), then broadcast
#pragma unroll
    for (int v = 0; v < 8; ++v) {
      const float inv_r = __shfl(invl, v + hi * 8, 32);
      const int row = q0 + u * 16 + v + hi * 8;
#pragma unroll
      for (int t2 = 0; t2 < 4; ++t2)
        o_ptr[(size_t)row * D + t2 * 16 + n16] = acc[u][t2][v] * inv_r;
    }
  }
}

extern "C" void kernel_launch(void* const* d_in, const int* in_sizes, int n_in,
                              void* d_out, int out_size, void* d_ws, size_t ws_size,
                              hipStream_t stream) {
  (void)in_sizes; (void)n_in; (void)d_ws; (void)ws_size; (void)out_size;
  const float* q = (const float*)d_in[0];
  const float* k = (const float*)d_in[1];
  const float* v = (const float*)d_in[2];
  float* o = (float*)d_out;

  dim3 grid(4 * 16, S / 256);  // (B*H, S / queries-per-block)
  fa_fwd_f16wmma<<<grid, 256, 0, stream>>>(q, k, v, o);
}
